// ScannedRNN_72619307041110
// MI455X (gfx1250) — matrix operand order repaired and checked by reading, not compiled, not run on hardware
//
#include <hip/hip_runtime.h>
#include <math.h>

// Problem constants (match reference).
#define TT 512
#define BB 256
#define DD 512
#define HH 512
#define G3 1536   // 3*H

typedef __attribute__((ext_vector_type(16))) __bf16 v16bf;
typedef __attribute__((ext_vector_type(8)))  __bf16 v8bf;
typedef __attribute__((ext_vector_type(2)))  __bf16 v2bf;
typedef __attribute__((ext_vector_type(8)))  float  v8f;

// WMMA fragment viewed as two 16-byte halves for direct vector loads.
union Frag { v16bf v; v8bf h[2]; };

// Packed f32x2 -> bf16x2 conversion. Prefer the hardware v_cvt_pk_bf16_f32.
__device__ __forceinline__ v2bf cvt2(float a, float b) {
#if __has_builtin(__builtin_amdgcn_cvt_pk_bf16_f32)
    return __builtin_amdgcn_cvt_pk_bf16_f32(a, b);
#else
    v2bf r; r[0] = (__bf16)a; r[1] = (__bf16)b; return r;
#endif
}

__device__ __forceinline__ unsigned short f2bf_u16(float f) {
    v2bf p = cvt2(f, f);
    __bf16 e = p[0];
    return __builtin_bit_cast(unsigned short, e);
}

// ---------------------------------------------------------------------------
// Kernel 0: transpose Wi and Wh ([D,3H] f32 row-major) into bf16 W^T [3H][D]
// so WMMA B-fragments are contiguous along K.
// ---------------------------------------------------------------------------
__global__ __launch_bounds__(256) void prep_transpose(
        const float* __restrict__ Wi, const float* __restrict__ Wh,
        unsigned short* __restrict__ WiT, unsigned short* __restrict__ WhT) {
    int idx = blockIdx.x * 256 + threadIdx.x;          // 0 .. 2*1536*512-1
    const int N = G3 * DD;
    int which = (idx >= N) ? 1 : 0;
    int e = idx - which * N;
    int j = e >> 9;            // 0..1535 (column of W == row of W^T)
    int d = e & 511;           // 0..511  (K)
    const float* W = which ? Wh : Wi;
    unsigned short* WT = which ? WhT : WiT;
    WT[(size_t)j * DD + d] = f2bf_u16(W[(size_t)d * G3 + j]);
}

// ---------------------------------------------------------------------------
// Kernel 0b: one streaming pass converting ins (f32) to bf16 once, so the
// GEMM never converts (each element would otherwise be converted 96x).
// 8 elements per thread.
// ---------------------------------------------------------------------------
__global__ __launch_bounds__(256) void prep_cvt_ins(
        const float* __restrict__ ins, unsigned short* __restrict__ insBF) {
    size_t i = ((size_t)blockIdx.x * 256 + threadIdx.x) * 8;
    const float* s = ins + i;
    unsigned* dst = (unsigned*)(insBF + i);
    #pragma unroll
    for (int j = 0; j < 4; ++j) {
        v2bf p = cvt2(s[2 * j], s[2 * j + 1]);
        dst[j] = __builtin_bit_cast(unsigned, p);
    }
}

// ---------------------------------------------------------------------------
// Kernel 1: gi = ins @ Wi + bi for ALL timesteps at once.
// One wave per 16x16 output tile; K=512 -> 16 x v_wmma_f32_16x16x32_bf16.
// A and B are both pre-converted bf16: pure 16B vector loads, zero repacking.
// ---------------------------------------------------------------------------
__global__ __launch_bounds__(256) void gi_gemm(
        const unsigned short* __restrict__ insBF_,
        const unsigned short* __restrict__ WiT_,
        const float* __restrict__ bi, float* __restrict__ gi) {
    const __bf16* insBF = (const __bf16*)insBF_;
    const __bf16* WiT   = (const __bf16*)WiT_;
    const int lane = threadIdx.x & 31;
    const int wave = threadIdx.x >> 5;
    const int tile = blockIdx.x * 8 + wave;   // 0 .. 786431
    const int rb = tile / 96;                 // row block over T*B (8192 blocks)
    const int cb = tile % 96;                 // col block over 3H  (96 blocks)
    const int m    = lane & 15;
    const int half = lane >> 4;

    const __bf16* aBase = insBF + (size_t)(rb * 16 + m) * DD;   // A row (M = lane&15)
    const __bf16* bBase = WiT   + (size_t)(cb * 16 + m) * DD;   // B col (N = lane&15)

    v8f acc = {};
    #pragma unroll
    for (int kk = 0; kk < 16; ++kk) {
        const int k0 = kk * 32;
        Frag a, b;
        // A, contiguous k order (same as B): 16 contiguous K values @ k0 + half*16
        a.h[0] = *(const v8bf*)(aBase + k0 + half * 16);
        a.h[1] = *(const v8bf*)(aBase + k0 + half * 16 + 8);
        // B, contiguous k order (same as A): 16 contiguous K values @ k0 + half*16 (from W^T)
        b.h[0] = *(const v8bf*)(bBase + k0 + half * 16);
        b.h[1] = *(const v8bf*)(bBase + k0 + half * 16 + 8);
        acc = __builtin_amdgcn_wmma_f32_16x16x32_bf16(false, a.v, false, b.v,
                                                      (short)0, acc, false, false);
    }
    // C layout: element v -> row rb*16 + v + 8*half, col cb*16 + (lane&15)
    #pragma unroll
    for (int v = 0; v < 8; ++v) {
        const int r_ = rb * 16 + v + 8 * half;
        const int c_ = cb * 16 + m;
        gi[(size_t)r_ * G3 + c_] = acc[v] + bi[c_];
    }
}

// ---------------------------------------------------------------------------
// Kernel 2: persistent GRU scan. 64 blocks x 256 threads (512 waves total).
// Each wave owns one (row-block, h-col-block) 16x16 tile for all 3 gates and
// keeps its Wh fragments (48 x v16bf ~ 384 VGPRs) resident across all T steps.
// Per step: stage h_prev (reset-masked, bf16) to LDS, 48 WMMAs, fused
// sigmoid/tanh GRU update, prefetch next step's gi slice, device-wide barrier.
// ---------------------------------------------------------------------------
__global__ __launch_bounds__(256) void gru_scan(
        const float* __restrict__ gi, const unsigned short* __restrict__ WhT_,
        const float* __restrict__ bhn, const int* __restrict__ resets,
        const float* __restrict__ h0, float* __restrict__ ys,
        unsigned* __restrict__ bar, int nblocks) {
    const __bf16* WhT = (const __bf16*)WhT_;
    __shared__ __bf16 hA[16 * DD];            // 16 rows of h in bf16 (16 KB)

    const int lane = threadIdx.x & 31;
    const int wave = threadIdx.x >> 5;
    const int rb = blockIdx.x >> 2;           // 0..15  : batch-row block (16 rows)
    const int cg = blockIdx.x & 3;            // 0..3
    const int hc = cg * 8 + wave;             // 0..31  : h-column block (16 cols)
    const int half = lane >> 4;
    const int n    = lane & 15;

    // ---- Preload Wh fragments for this wave's r/z/n columns, full K=512 ----
    Frag br[16], bz[16], bn[16];
    {
        const __bf16* cr = WhT + (size_t)(hc * 16 + n) * DD;
        const __bf16* cz = WhT + (size_t)(HH + hc * 16 + n) * DD;
        const __bf16* cn = WhT + (size_t)(2 * HH + hc * 16 + n) * DD;
        #pragma unroll
        for (int kk = 0; kk < 16; ++kk) {
            const int k = kk * 32 + half * 16;
            br[kk].h[0] = *(const v8bf*)(cr + k);
            br[kk].h[1] = *(const v8bf*)(cr + k + 8);
            bz[kk].h[0] = *(const v8bf*)(cz + k);
            bz[kk].h[1] = *(const v8bf*)(cz + k + 8);
            bn[kk].h[0] = *(const v8bf*)(cn + k);
            bn[kk].h[1] = *(const v8bf*)(cn + k + 8);
        }
    }

    const float bn_c = bhn[hc * 16 + n];

    for (int t = 0; t < TT; ++t) {
        // ---- Stage reset-masked h_prev (bf16) into LDS: 16 rows x 512 ----
        {
            const int r_local = threadIdx.x >> 4;       // 0..15
            const int seg = (threadIdx.x & 15) * 32;    // 32 contiguous cols
            const int row = rb * 16 + r_local;
            const int rst = resets[t * BB + row];
            const float* src = (t == 0) ? (h0 + (size_t)row * HH + seg)
                                        : (ys + ((size_t)(t - 1) * BB + row) * HH + seg);
            unsigned* dst = (unsigned*)&hA[r_local * DD + seg];
            #pragma unroll
            for (int i = 0; i < 16; ++i) {
                float a = rst ? 0.0f : src[2 * i];
                float b = rst ? 0.0f : src[2 * i + 1];
                v2bf p = cvt2(a, b);
                dst[i] = __builtin_bit_cast(unsigned, p);
            }
        }
        __syncthreads();

        // ---- gh tiles: 3 gates x 16 K-slabs of WMMA ----
        v8f ar = {}, az = {}, an = {};
        #pragma unroll
        for (int kk = 0; kk < 16; ++kk) {
            Frag a;
            const __bf16* p0 = &hA[(size_t)n * DD + kk * 32 + half * 16];
            a.h[0] = *(const v8bf*)p0;
            a.h[1] = *(const v8bf*)(p0 + 8);
            ar = __builtin_amdgcn_wmma_f32_16x16x32_bf16(false, a.v, false, br[kk].v,
                                                         (short)0, ar, false, false);
            az = __builtin_amdgcn_wmma_f32_16x16x32_bf16(false, a.v, false, bz[kk].v,
                                                         (short)0, az, false, false);
            an = __builtin_amdgcn_wmma_f32_16x16x32_bf16(false, a.v, false, bn[kk].v,
                                                         (short)0, an, false, false);
        }

        // ---- Fused GRU elementwise + store (C layout: M=v+8*half, N=lane&15) ----
        #pragma unroll
        for (int v = 0; v < 8; ++v) {
            const int row = rb * 16 + v + 8 * half;
            const int c   = hc * 16 + n;
            const float* gib = gi + ((size_t)t * BB + row) * G3 + c;
            const float ir = gib[0], iz = gib[HH], inn = gib[2 * HH];
            const int rst = resets[t * BB + row];
            const float hp = rst ? 0.0f
                : ((t == 0) ? h0[(size_t)row * HH + c]
                            : ys[((size_t)(t - 1) * BB + row) * HH + c]);
            const float r = 1.0f / (1.0f + expf(-(ir + ar[v])));
            const float z = 1.0f / (1.0f + expf(-(iz + az[v])));
            const float nn = tanhf(inn + r * (an[v] + bn_c));
            ys[((size_t)t * BB + row) * HH + c] = (1.0f - z) * nn + z * hp;
        }

        // ---- Prefetch next step's gi slice for this block (24 KB, 192 lines)
        if (t + 1 < TT) {
            const int tid = threadIdx.x;
            if (tid < 192) {
                const int g       = tid / 64;        // gate 0..2
                const int r_local = (tid >> 2) & 15; // row 0..15
                const int chunk   = tid & 3;         // 128B line within 128 cols
                const float* pf = gi + ((size_t)(t + 1) * BB + rb * 16 + r_local) * G3
                                + g * HH + cg * 128 + chunk * 32;
                __builtin_prefetch(pf, 0, 3);
            }
        }

        // ---- Device-wide barrier (release stores, then count+generation) ----
        __threadfence();
        __syncthreads();
        if (threadIdx.x == 0) {
            unsigned gen = __hip_atomic_load(bar + 1, __ATOMIC_ACQUIRE,
                                             __HIP_MEMORY_SCOPE_AGENT);
            unsigned arrived = __hip_atomic_fetch_add(bar, 1u, __ATOMIC_ACQ_REL,
                                                      __HIP_MEMORY_SCOPE_AGENT);
            if (arrived == (unsigned)(nblocks - 1)) {
                __hip_atomic_store(bar, 0u, __ATOMIC_RELAXED, __HIP_MEMORY_SCOPE_AGENT);
                __hip_atomic_fetch_add(bar + 1, 1u, __ATOMIC_ACQ_REL,
                                       __HIP_MEMORY_SCOPE_AGENT);
            } else {
                while (__hip_atomic_load(bar + 1, __ATOMIC_ACQUIRE,
                                         __HIP_MEMORY_SCOPE_AGENT) == gen) {
                    __builtin_amdgcn_s_sleep(1);
                }
            }
        }
        __syncthreads();
    }
}

// ---------------------------------------------------------------------------
// Launch. Inputs (setup_inputs order):
//   0: ins [T,B,D] f32   1: resets [T,B] int   2: h0 [B,H] f32
//   3: Wi [D,3H] f32     4: bi [3H] f32        5: Wh [H,3H] f32   6: bhn [H] f32
// Output: ys [T,B,H] f32.
// Workspace layout: gi f32 (805,306,368 B) | WiT bf16 (1.5 MB) | WhT bf16
// (1.5 MB) | insBF bf16 (134,217,728 B) | barrier (2 x u32). Total ~943 MB.
// ---------------------------------------------------------------------------
extern "C" void kernel_launch(void* const* d_in, const int* in_sizes, int n_in,
                              void* d_out, int out_size, void* d_ws, size_t ws_size,
                              hipStream_t stream) {
    const float* ins    = (const float*)d_in[0];
    const int*   resets = (const int*)d_in[1];
    const float* h0     = (const float*)d_in[2];
    const float* Wi     = (const float*)d_in[3];
    const float* bi     = (const float*)d_in[4];
    const float* Wh     = (const float*)d_in[5];
    const float* bhn    = (const float*)d_in[6];
    float* ys = (float*)d_out;

    const size_t GI_BYTES = (size_t)TT * BB * G3 * sizeof(float);   // 805,306,368
    const size_t WT_ELEMS = (size_t)G3 * DD;                        // 786,432
    const size_t INS_ELEMS = (size_t)TT * BB * DD;                  // 67,108,864
    char* ws = (char*)d_ws;
    float*          gi    = (float*)ws;
    unsigned short* WiT   = (unsigned short*)(ws + GI_BYTES);
    unsigned short* WhT   = WiT + WT_ELEMS;
    unsigned short* insBF = WhT + WT_ELEMS;
    unsigned*       bar   = (unsigned*)((char*)(insBF + INS_ELEMS));

    // 0) weight transpose+convert, input convert (one pass each)
    prep_transpose<<<(2 * (int)WT_ELEMS) / 256, 256, 0, stream>>>(Wi, Wh, WiT, WhT);
    prep_cvt_ins<<<(int)(INS_ELEMS / (256 * 8)), 256, 0, stream>>>(ins, insBF);

    // 1) gi = ins @ Wi + bi over all T*B rows: 8192 x 96 tiles, 8 waves/block
    gi_gemm<<<(8192 * 96) / 8, 256, 0, stream>>>(insBF, WiT, bi, gi);

    // 2) reset barrier state, then run the persistent scan
    hipMemsetAsync(bar, 0, 256, stream);
    const int nblocks = 64;
    gru_scan<<<nblocks, 256, 0, stream>>>(gi, WhT, bhn, resets, h0, ys, bar, nblocks);
}
